// GenericAttentionBlock_4552665334361
// MI455X (gfx1250) — matrix operand order repaired and checked by reading, not compiled, not run on hardware
//
#include <hip/hip_runtime.h>
#include <math.h>

// ---------------------------------------------------------------------------
// MI455X (gfx1250) attention block.
// Precision: f32 split into (hi,lo) _Float16 pairs; every GEMM does
// 3x v_wmma_f32_16x16x32_f16 per K-step (hh + hl + lh) => ~f32 accuracy at
// matrix-core rate. Attention is flash-style (no LxL materialization).
// Tiling: GEMM waves own 32x64 macro-tiles (24 WMMAs per 24 b128 loads);
// attention waves own 32 queries so K/V fragments feed 48 WMMAs per 4KB.
// ---------------------------------------------------------------------------

#define D_MODEL 1024
#define D_HEAD  64
#define N_HEADS 16
#define B_SZ    2
#define SEQ     2048
#define NTOK    (B_SZ * SEQ)      // 4096 tokens
#define EPS_F   1e-6f

typedef _Float16 v8h  __attribute__((ext_vector_type(8)));
typedef _Float16 v16h __attribute__((ext_vector_type(16)));
typedef float    v8f  __attribute__((ext_vector_type(8)));

__device__ __forceinline__ v8f wmma_step(v16h a, v16h b, v8f c) {
  // D = A(16x32 f16) * B(32x16 f16) + C(16x16 f32)
  return __builtin_amdgcn_wmma_f32_16x16x32_f16(
      /*neg_a=*/false, a, /*neg_b=*/false, b,
      /*c_mod=*/(short)0, c, /*reuse_a=*/false, /*reuse_b=*/false);
}

__device__ __forceinline__ v16h cat8(v8h lo, v8h hi) {
  return __builtin_shufflevector(lo, hi, 0, 1, 2, 3, 4, 5, 6, 7,
                                 8, 9, 10, 11, 12, 13, 14, 15);
}

// A-matrix 16x32 f16 fragment from row-major [16 x stride] array at K offset k0.
// Lane l<16: row l, K = k0+0..7 / k0+16..23 ; lane l>=16: row l-16, +8.. / +24..
__device__ __forceinline__ v16h ld_frag_a(const _Float16* base, int stride,
                                          int lane, int k0) {
  const int row  = lane & 15;
  const int koff = k0 + ((lane & 16) ? 8 : 0);
  const _Float16* p = base + (size_t)row * stride + koff;
  return cat8(*(const v8h*)p, *(const v8h*)(p + 16));
}

// B-matrix 32x16 f16 fragment. Array is [N x stride]; array-row n is B column
// n (K contiguous). Lane l<16: col l, K = k0+0..7 / k0+16..23 ; lane l>=16: +8.. / +24..
__device__ __forceinline__ v16h ld_frag_b(const _Float16* base, int stride,
                                          int lane, int k0) {
  const int col  = lane & 15;
  const int koff = k0 + ((lane & 16) ? 8 : 0);
  const _Float16* p = base + (size_t)col * stride + koff;
  return cat8(*(const v8h*)p, *(const v8h*)(p + 16));
}

__device__ __forceinline__ void split_store(_Float16* hi, _Float16* lo,
                                            size_t idx, float v) {
  _Float16 h = (_Float16)v;
  hi[idx] = h;
  lo[idx] = (_Float16)(v - (float)h);
}

// ---------------------------------------------------------------------------
// Elementwise f32 -> (hi,lo) f16 split (weights).
// ---------------------------------------------------------------------------
__global__ void __launch_bounds__(256)
split_f32_f16(const float* __restrict__ w, _Float16* __restrict__ hi,
              _Float16* __restrict__ lo, int n) {
  int i = blockIdx.x * 256 + threadIdx.x;
  if (i < n) split_store(hi, lo, (size_t)i, w[i]);
}

// ---------------------------------------------------------------------------
// RMSNorm + split. One block per token row (1024 elems, 256 threads x 4).
// ---------------------------------------------------------------------------
__global__ void __launch_bounds__(256)
rmsnorm_split(const float* __restrict__ x, const float* __restrict__ gamma,
              _Float16* __restrict__ hhi, _Float16* __restrict__ hlo) {
  __shared__ float red[8];
  const int row = blockIdx.x, tid = threadIdx.x;
  const float* xr = x + (size_t)row * D_MODEL;
  float v[4];
  float ssq = 0.f;
#pragma unroll
  for (int i = 0; i < 4; ++i) {
    v[i] = xr[tid + 256 * i];
    ssq += v[i] * v[i];
  }
#pragma unroll
  for (int mk = 1; mk <= 16; mk <<= 1) ssq += __shfl_xor(ssq, mk, 32);
  if ((tid & 31) == 0) red[tid >> 5] = ssq;
  __syncthreads();
  float tot = 0.f;
#pragma unroll
  for (int i = 0; i < 8; ++i) tot += red[i];
  const float r = rsqrtf(tot * (1.0f / D_MODEL) + EPS_F);
#pragma unroll
  for (int i = 0; i < 4; ++i) {
    int c = tid + 256 * i;
    split_store(hhi, hlo, (size_t)row * D_MODEL + c, v[i] * gamma[c] * r);
  }
}

// ---------------------------------------------------------------------------
// Split-f16 GEMM: C[M,N] = A[M,K] @ B[N,K]^T (+ optional skip).
// 8 waves/block; each wave owns a 32x64 macro-tile (2x4 WMMA C tiles).
// Per 32-wide K-step: 24 b128 loads feed 24 WMMAs (~128B per WMMA).
// Requires M % 32 == 0, N % 64 == 0 (true for all call sites).
// ---------------------------------------------------------------------------
__global__ void __launch_bounds__(256)
gemm_sf16(const _Float16* __restrict__ Ah, const _Float16* __restrict__ Al,
          const _Float16* __restrict__ Bh, const _Float16* __restrict__ Bl,
          float* __restrict__ C, const float* __restrict__ skip,
          int M, int N, int K) {
  const int wid = threadIdx.x >> 5, lane = threadIdx.x & 31;
  const int tiles_n = N >> 6;
  const int tile = blockIdx.x * 8 + wid;
  if (tile >= (M >> 5) * tiles_n) return;  // wave-uniform
  const int tm = tile / tiles_n, tn = tile % tiles_n;

  const _Float16* ah = Ah + (size_t)tm * 32 * K;
  const _Float16* al = Al + (size_t)tm * 32 * K;
  const _Float16* bh = Bh + (size_t)tn * 64 * K;
  const _Float16* bl = Bl + (size_t)tn * 64 * K;

  const v8f vz = {0.f, 0.f, 0.f, 0.f, 0.f, 0.f, 0.f, 0.f};
  v8f acc[2][4];
#pragma unroll
  for (int mi = 0; mi < 2; ++mi)
#pragma unroll
    for (int ni = 0; ni < 4; ++ni) acc[mi][ni] = vz;

  for (int k0 = 0; k0 < K; k0 += 32) {
    v16h fah[2], fal[2];
#pragma unroll
    for (int mi = 0; mi < 2; ++mi) {
      fah[mi] = ld_frag_a(ah + (size_t)mi * 16 * K, K, lane, k0);
      fal[mi] = ld_frag_a(al + (size_t)mi * 16 * K, K, lane, k0);
    }
#pragma unroll
    for (int ni = 0; ni < 4; ++ni) {
      v16h fbh = ld_frag_b(bh + (size_t)ni * 16 * K, K, lane, k0);
      v16h fbl = ld_frag_b(bl + (size_t)ni * 16 * K, K, lane, k0);
#pragma unroll
      for (int mi = 0; mi < 2; ++mi) {
        acc[mi][ni] = wmma_step(fah[mi], fbh, acc[mi][ni]);
        acc[mi][ni] = wmma_step(fah[mi], fbl, acc[mi][ni]);
        acc[mi][ni] = wmma_step(fal[mi], fbh, acc[mi][ni]);
      }
    }
  }

  const int prow = (lane & 16) ? 8 : 0;
  const int pcol = lane & 15;
#pragma unroll
  for (int mi = 0; mi < 2; ++mi) {
    const int row0 = tm * 32 + mi * 16 + prow;
#pragma unroll
    for (int ni = 0; ni < 4; ++ni) {
      const int col = tn * 64 + ni * 16 + pcol;
#pragma unroll
      for (int r = 0; r < 8; ++r) {
        size_t idx = (size_t)(row0 + r) * N + col;
        float v = acc[mi][ni][r];
        if (skip) v += skip[idx];
        C[idx] = v;
      }
    }
  }
}

// ---------------------------------------------------------------------------
// QKV post-processing: per (b,l,h) row of 64 — cosine-sim scaling of q/k,
// 3D RoPE, f16 split. Q/K stored [b,h,l,64]; V stored transposed [b,h,64,l]
// so attention's B-fragments are contiguous. One wave per row, 8 rows/block.
// ---------------------------------------------------------------------------
__global__ void __launch_bounds__(256)
qkv_post(const float* __restrict__ qkv, const float* __restrict__ pos,
         const float* __restrict__ attn_scale,
         _Float16* __restrict__ qh, _Float16* __restrict__ ql,
         _Float16* __restrict__ kh, _Float16* __restrict__ kl,
         _Float16* __restrict__ vh, _Float16* __restrict__ vl) {
  __shared__ float sq[8][64];
  __shared__ float sk[8][64];
  const int wid = threadIdx.x >> 5, lane = threadIdx.x & 31;
  const int idx = blockIdx.x * 8 + wid;  // token*16 + head
  const int hh = idx & 15;
  const int token = idx >> 4;
  const int b = token >> 11;
  const int l = token & 2047;

  const float* base = qkv + (size_t)token * (3 * D_MODEL) + hh * 64;
  float q0 = base[lane], q1 = base[lane + 32];
  float k0 = base[1024 + lane], k1 = base[1024 + lane + 32];
  float v0 = base[2048 + lane], v1 = base[2048 + lane + 32];

  float ssq = q0 * q0 + q1 * q1, ssk = k0 * k0 + k1 * k1;
#pragma unroll
  for (int mk = 1; mk <= 16; mk <<= 1) {
    ssq += __shfl_xor(ssq, mk, 32);
    ssk += __shfl_xor(ssk, mk, 32);
  }
  const float sc = sqrtf(attn_scale[hh]);
  const float rq = sc * rsqrtf(ssq + EPS_F);
  const float rk = sc * rsqrtf(ssk + EPS_F);
  q0 *= rq; q1 *= rq; k0 *= rk; k1 *= rk;
  sq[wid][lane] = q0; sq[wid][lane + 32] = q1;
  sk[wid][lane] = k0; sk[wid][lane + 32] = k1;
  asm volatile("s_wait_dscnt 0" ::: "memory");  // wave-internal LDS ordering

  float pp[3];
#pragma unroll
  for (int i = 0; i < 3; ++i) pp[i] = pos[(size_t)token * 3 + i];

  const size_t qbase  = ((size_t)(b * N_HEADS + hh) * SEQ + l) * 64;
  const size_t vbase0 = ((size_t)(b * N_HEADS + hh) * 64) * SEQ + l;

  // freqs[h][j] = exp(log(pi) + (j*16 + h) * ln(10)/160)
#pragma unroll
  for (int t = 0; t < 2; ++t) {
    const int dd = lane + 32 * t;
    float yq, yk;
    if (dd < 60) {
      const int j  = (dd < 30) ? dd : dd - 30;
      const int fi = j / 10, fj = j - 10 * fi;
      const float fr = expf(1.14472988584940017f +
                            (float)(fj * 16 + hh) * 0.01439115683121279f);
      const float th = pp[fi] * fr;
      const float cth = cosf(th), sth = sinf(th);
      if (dd < 30) {
        yq = sq[wid][dd] * cth - sq[wid][dd + 30] * sth;
        yk = sk[wid][dd] * cth - sk[wid][dd + 30] * sth;
      } else {
        yq = sq[wid][dd] * cth + sq[wid][j] * sth;
        yk = sk[wid][dd] * cth + sk[wid][j] * sth;
      }
    } else {
      yq = sq[wid][dd];
      yk = sk[wid][dd];
    }
    split_store(qh, ql, qbase + dd, yq);
    split_store(kh, kl, qbase + dd, yk);
    split_store(vh, vl, vbase0 + (size_t)dd * SEQ, t ? v1 : v0);
  }
}

// ---------------------------------------------------------------------------
// Flash attention. One wave per (b, h, 32-query tile); 32 keys per iteration.
// The two 16-query subtiles share every K and V fragment, so each 32-key
// iteration's 32 b128 loads feed 48 WMMAs. P is re-swizzled from C-layout
// to A-layout through per-wave LDS (wave-internal s_wait_dscnt; LDS ops from
// one wave execute in order, so next iteration's stores cannot pass reads).
// ---------------------------------------------------------------------------
__global__ void __launch_bounds__(256)
attn_flash(const _Float16* __restrict__ qh, const _Float16* __restrict__ ql,
           const _Float16* __restrict__ kh, const _Float16* __restrict__ kl,
           const _Float16* __restrict__ vth, const _Float16* __restrict__ vtl,
           _Float16* __restrict__ oh, _Float16* __restrict__ ol) {
  __shared__ _Float16 pbh[8][2 * 16 * 32];
  __shared__ _Float16 pbl[8][2 * 16 * 32];
  const int wid = threadIdx.x >> 5, lane = threadIdx.x & 31;
  const int task = blockIdx.x * 8 + wid;  // 2048 tasks total
  const int qt = task & 63;               // 32-query tile
  const int bh = task >> 6;               // 0..31
  const int b = bh >> 4, hh = bh & 15;

  const _Float16* qhb = qh + ((size_t)bh * SEQ + qt * 32) * 64;
  const _Float16* qlb = ql + ((size_t)bh * SEQ + qt * 32) * 64;
  const _Float16* khb = kh + (size_t)bh * SEQ * 64;
  const _Float16* klb = kl + (size_t)bh * SEQ * 64;
  const _Float16* vhb = vth + (size_t)bh * 64 * SEQ;
  const _Float16* vlb = vtl + (size_t)bh * 64 * SEQ;

  v16h aqh[2][2], aql[2][2];  // [q-subtile][K-step]
#pragma unroll
  for (int q = 0; q < 2; ++q)
#pragma unroll
    for (int s = 0; s < 2; ++s) {
      aqh[q][s] = ld_frag_a(qhb + (size_t)q * 16 * 64, 64, lane, s * 32);
      aql[q][s] = ld_frag_a(qlb + (size_t)q * 16 * 64, 64, lane, s * 32);
    }

  const v8f vz = {0.f, 0.f, 0.f, 0.f, 0.f, 0.f, 0.f, 0.f};
  v8f oacc[2][4];
#pragma unroll
  for (int q = 0; q < 2; ++q)
#pragma unroll
    for (int c = 0; c < 4; ++c) oacc[q][c] = vz;
  float m_i[2][8], l_i[2][8];
#pragma unroll
  for (int q = 0; q < 2; ++q)
#pragma unroll
    for (int r = 0; r < 8; ++r) { m_i[q][r] = -3.0e38f; l_i[q][r] = 0.f; }

  _Float16* ph = pbh[wid];
  _Float16* pl = pbl[wid];
  const int prow = (lane & 16) ? 8 : 0;
  const int pcol = lane & 15;

  for (int j = 0; j < SEQ; j += 32) {
    // ---- scores for keys j..j+31, both query subtiles ----------------
    v8f s_t[2][2];  // [q-subtile][key-subtile]
#pragma unroll
    for (int q = 0; q < 2; ++q) { s_t[q][0] = vz; s_t[q][1] = vz; }
#pragma unroll
    for (int s = 0; s < 2; ++s) {
#pragma unroll
      for (int t = 0; t < 2; ++t) {
        v16h kb_h = ld_frag_b(khb + (size_t)(j + 16 * t) * 64, 64, lane, s * 32);
        v16h kb_l = ld_frag_b(klb + (size_t)(j + 16 * t) * 64, 64, lane, s * 32);
#pragma unroll
        for (int q = 0; q < 2; ++q) {
          s_t[q][t] = wmma_step(aqh[q][s], kb_h, s_t[q][t]);
          s_t[q][t] = wmma_step(aqh[q][s], kb_l, s_t[q][t]);
          s_t[q][t] = wmma_step(aql[q][s], kb_h, s_t[q][t]);
        }
      }
    }
    // ---- online softmax (rows live in 16-lane halves) ----------------
#pragma unroll
    for (int q = 0; q < 2; ++q) {
      float alpha[8];
#pragma unroll
      for (int r = 0; r < 8; ++r) {
        float t = fmaxf(s_t[q][0][r], s_t[q][1][r]);
#pragma unroll
        for (int mk = 1; mk <= 8; mk <<= 1) t = fmaxf(t, __shfl_xor(t, mk, 32));
        const float mn = fmaxf(m_i[q][r], t);
        alpha[r] = expf(m_i[q][r] - mn);
        m_i[q][r] = mn;
        const float p0 = expf(s_t[q][0][r] - mn);
        const float p1 = expf(s_t[q][1][r] - mn);
        s_t[q][0][r] = p0;
        s_t[q][1][r] = p1;
        float rs = p0 + p1;
#pragma unroll
        for (int mk = 1; mk <= 8; mk <<= 1) rs += __shfl_xor(rs, mk, 32);
        l_i[q][r] = l_i[q][r] * alpha[r] + rs;
      }
#pragma unroll
      for (int c = 0; c < 4; ++c)
#pragma unroll
        for (int r = 0; r < 8; ++r) oacc[q][c][r] *= alpha[r];

      // ---- P: C-layout -> LDS (row-major 16x32, hi/lo split) ---------
#pragma unroll
      for (int r = 0; r < 8; ++r) {
        const int a0 = q * 512 + (prow + r) * 32 + pcol;
        const float p0 = s_t[q][0][r];
        _Float16 h0 = (_Float16)p0;
        ph[a0] = h0;
        pl[a0] = (_Float16)(p0 - (float)h0);
        const float p1 = s_t[q][1][r];
        _Float16 h1 = (_Float16)p1;
        ph[a0 + 16] = h1;
        pl[a0 + 16] = (_Float16)(p1 - (float)h1);
      }
    }
    asm volatile("s_wait_dscnt 0" ::: "memory");
    v16h pah[2], pal[2];
#pragma unroll
    for (int q = 0; q < 2; ++q) {
      pah[q] = ld_frag_a(ph + q * 512, 32, lane, 0);
      pal[q] = ld_frag_a(pl + q * 512, 32, lane, 0);
    }
    // ---- O += P @ V (V^T layout: dim-major, keys contiguous) ---------
#pragma unroll
    for (int c = 0; c < 4; ++c) {
      v16h vb_h = ld_frag_b(vhb + (size_t)c * 16 * SEQ, SEQ, lane, j);
      v16h vb_l = ld_frag_b(vlb + (size_t)c * 16 * SEQ, SEQ, lane, j);
#pragma unroll
      for (int q = 0; q < 2; ++q) {
        oacc[q][c] = wmma_step(pah[q], vb_h, oacc[q][c]);
        oacc[q][c] = wmma_step(pah[q], vb_l, oacc[q][c]);
        oacc[q][c] = wmma_step(pal[q], vb_h, oacc[q][c]);
      }
    }
  }

  // ---- finalize: divide by row sum, split-store as [token, d_model] --
#pragma unroll
  for (int q = 0; q < 2; ++q)
#pragma unroll
    for (int r = 0; r < 8; ++r) {
      const float inv = 1.0f / l_i[q][r];
      const int tok = b * SEQ + qt * 32 + q * 16 + prow + r;
#pragma unroll
      for (int c = 0; c < 4; ++c) {
        const size_t idx = (size_t)tok * D_MODEL + hh * 64 + c * 16 + pcol;
        split_store(oh, ol, idx, oacc[q][c][r] * inv);
      }
    }
}

// ---------------------------------------------------------------------------
// Host-side orchestration.
// ---------------------------------------------------------------------------
extern "C" void kernel_launch(void* const* d_in, const int* in_sizes, int n_in,
                              void* d_out, int out_size, void* d_ws,
                              size_t ws_size, hipStream_t stream) {
  (void)in_sizes; (void)n_in; (void)out_size; (void)ws_size;
  const float* x      = (const float*)d_in[0];
  const float* pos    = (const float*)d_in[1];
  const float* nscale = (const float*)d_in[2];
  const float* w_qkv  = (const float*)d_in[3];
  const float* w_out  = (const float*)d_in[4];
  const float* ascale = (const float*)d_in[5];
  float* out = (float*)d_out;

  char* ws = (char*)d_ws;
  size_t off = 0;
  auto take = [&](size_t n) {
    size_t r = off;
    off += (n + 255) & ~(size_t)255;
    return r;
  };
  const size_t SZ16 = (size_t)NTOK * D_MODEL * 2;  // 8 MiB of f16
  const size_t oHHI = take(SZ16);                  // h_hi   (reused as q_hi)
  const size_t oHLO = take(SZ16);                  // h_lo   (reused as q_lo)
  const size_t oWQH = take((size_t)3 * D_MODEL * D_MODEL * 2);
  const size_t oWQL = take((size_t)3 * D_MODEL * D_MODEL * 2);
  const size_t oWOH = take((size_t)D_MODEL * D_MODEL * 2);
  const size_t oWOL = take((size_t)D_MODEL * D_MODEL * 2);
  const size_t oQKV = take((size_t)NTOK * 3 * D_MODEL * 4);  // 48 MiB f32
  const size_t oKHI = take(SZ16);
  const size_t oKLO = take(SZ16);
  const size_t oVHI = take(SZ16);
  const size_t oVLO = take(SZ16);
  // overlaps (strictly sequential producers/consumers):
  const size_t oQHI = oHHI, oQLO = oHLO;           // h dead after QKV GEMM
  const size_t oOHI = oQKV, oOLO = oQKV + SZ16;    // qkv dead after qkv_post

  _Float16* hhi = (_Float16*)(ws + oHHI);
  _Float16* hlo = (_Float16*)(ws + oHLO);
  _Float16* wqh = (_Float16*)(ws + oWQH);
  _Float16* wql = (_Float16*)(ws + oWQL);
  _Float16* woh = (_Float16*)(ws + oWOH);
  _Float16* wol = (_Float16*)(ws + oWOL);
  float*    qkv = (float*)(ws + oQKV);
  _Float16* qhi = (_Float16*)(ws + oQHI);
  _Float16* qlo = (_Float16*)(ws + oQLO);
  _Float16* khi = (_Float16*)(ws + oKHI);
  _Float16* klo = (_Float16*)(ws + oKLO);
  _Float16* vhi = (_Float16*)(ws + oVHI);
  _Float16* vlo = (_Float16*)(ws + oVLO);
  _Float16* ohi = (_Float16*)(ws + oOHI);
  _Float16* olo = (_Float16*)(ws + oOLO);

  // 1) weight splits
  {
    int nq = 3 * D_MODEL * D_MODEL, no = D_MODEL * D_MODEL;
    split_f32_f16<<<(nq + 255) / 256, 256, 0, stream>>>(w_qkv, wqh, wql, nq);
    split_f32_f16<<<(no + 255) / 256, 256, 0, stream>>>(w_out, woh, wol, no);
  }
  // 2) RMSNorm + split
  rmsnorm_split<<<NTOK, 256, 0, stream>>>(x, nscale, hhi, hlo);
  // 3) QKV GEMM: (4096 x 3072) = h @ w_qkv^T ; 32x64 macro-tiles
  gemm_sf16<<<(NTOK / 32) * (3 * D_MODEL / 64) / 8, 256, 0, stream>>>(
      hhi, hlo, wqh, wql, qkv, nullptr, NTOK, 3 * D_MODEL, D_MODEL);
  // 4) cosine-sim scaling + RoPE + split; V transposed
  qkv_post<<<NTOK * N_HEADS / 8, 256, 0, stream>>>(qkv, pos, ascale, qhi, qlo,
                                                   khi, klo, vhi, vlo);
  // 5) flash attention (32 queries per wave)
  attn_flash<<<(B_SZ * N_HEADS * (SEQ / 32)) / 8, 256, 0, stream>>>(
      qhi, qlo, khi, klo, vhi, vlo, ohi, olo);
  // 6) out projection + skip ; 32x64 macro-tiles
  gemm_sf16<<<(NTOK / 32) * (D_MODEL / 64) / 8, 256, 0, stream>>>(
      ohi, olo, woh, wol, out, x, NTOK, D_MODEL, D_MODEL);
}
